// CategoryRouter_6416681140465
// MI455X (gfx1250) — matrix operand order repaired and checked by reading, not compiled, not run on hardware
//
#include <hip/hip_runtime.h>
#include <hip/hip_bf16.h>
#include <stdint.h>

// Problem constants (match reference)
#define BATCH 2
#define SEQ   512
#define NTOK  (BATCH*SEQ)   // 1024
#define DDIM  768
#define FDIM  3072
#define VDIM  1024
#define NROUTE 8

typedef __bf16 bf16_t;
typedef __attribute__((ext_vector_type(16))) __bf16 v16bf;
typedef __attribute__((ext_vector_type(8)))  float  v8f;

// ---------------- workspace layout (bytes) ----------------
// [0,36)       : offs[9]   prefix offsets per route
// [64,4160)    : list[1024] gathered token ids (route-sorted)
// [8192,+1.5M) : Xg  bf16 [1024][768]   gathered+converted tokens
// [~1.58M,+6M) : H   bf16 [1024][3072]  hidden activations
#define WS_OFFS 0
#define WS_LIST 64
#define WS_XG   8192
#define WS_H    (WS_XG + NTOK*DDIM*2)

// ============ kernel 1: build route-sorted token lists ============
__global__ __launch_bounds__(256) void cr_build_lists(
    const int* __restrict__ route_ids, int* __restrict__ offs, int* __restrict__ list) {
  __shared__ int cnt[NROUTE];
  __shared__ int base[NROUTE];
  __shared__ int cur[NROUTE];
  const int t = threadIdx.x;
  if (t < NROUTE) cnt[t] = 0;
  __syncthreads();
  for (int i = t; i < NTOK; i += 256) atomicAdd(&cnt[route_ids[i]], 1);
  __syncthreads();
  if (t == 0) {
    int acc = 0;
    for (int r = 0; r < NROUTE; ++r) { base[r] = acc; cur[r] = acc; acc += cnt[r]; }
  }
  __syncthreads();
  for (int i = t; i < NTOK; i += 256) {
    int r = route_ids[i];
    int slot = atomicAdd(&cur[r], 1);
    list[slot] = i;
  }
  if (t < NROUTE) offs[t] = base[t];
  if (t == 0)     offs[NROUTE] = NTOK;
}

// ============ kernel 2: gather tokens + fp32 -> bf16 (vectorized) ============
__global__ __launch_bounds__(192) void cr_gather_bf16(
    const float* __restrict__ e, const int* __restrict__ list, bf16_t* __restrict__ Xg) {
  const int g = blockIdx.x;
  const int tok = list[g];
  const float* __restrict__ src = e + (size_t)tok * DDIM;
  bf16_t* __restrict__ dst = Xg + (size_t)g * DDIM;
  const int d = threadIdx.x * 4;            // 192 threads * 4 = 768
  float4 f = *(const float4*)(src + d);
  union { bf16_t h[4]; uint2 u; } pk;
  pk.h[0] = (bf16_t)f.x; pk.h[1] = (bf16_t)f.y;
  pk.h[2] = (bf16_t)f.z; pk.h[3] = (bf16_t)f.w;
  *(uint2*)(dst + d) = pk.u;
}

// ============ grouped GEMM, bf16 WMMA, double-buffered LDS ============
// Block: 256 threads (8 waves), tile 64(M) x 128(N), BK=32.
// Waves: 4(M) x 2(N); each wave: one 16x32 A fragment vs four 16x16 N-subtiles.
// LDS A: row-major [64][40]; LDS B: column-major [128][40]  (WMMA-layout friendly).
// Pipeline per K-step: issue next tile's global loads -> WMMA on current buffer
// -> convert+store next tile -> single barrier -> swap.
template<int KDIM, int NDIM, bool RELU_BF16_OUT>
__global__ __launch_bounds__(256) void cr_gemm_routed(
    const bf16_t* __restrict__ A,     // [1024][KDIM] route-sorted rows
    const float*  __restrict__ W,     // [R][KDIM][NDIM] fp32 weights
    const float*  __restrict__ bias,  // [R][NDIM]
    const int*    __restrict__ offs,  // [R+1]
    const int*    __restrict__ list,  // [1024] scatter map (used when !RELU)
    bf16_t* __restrict__ outH,        // hidden out (RELU path), [1024][NDIM]
    float*  __restrict__ outL) {      // logits out (final path), scattered [tok][NDIM]
  constexpr int BM = 64, BN = 128, BK = 32, LDP = BK + 8;  // pad = 16B
  const int r    = blockIdx.z;
  const int row0 = offs[r];
  const int row1 = offs[r + 1];
  const int mbase = row0 + blockIdx.y * BM;
  if (mbase >= row1) return;
  const int nbase = blockIdx.x * BN;

  __shared__ bf16_t sA[2][BM][LDP];
  __shared__ bf16_t sB[2][BN][LDP];

  const int tid  = threadIdx.x;
  const int lane = tid & 31;
  const int wid  = tid >> 5;
  const int wm   = wid & 3;        // M sub-block (x16)
  const int wn   = wid >> 2;       // N half (x64)
  const int half = lane >> 4;
  const int l15  = lane & 15;

  const float* __restrict__ Wr = W + (size_t)r * KDIM * NDIM;

  // staging maps (fixed per thread)
  const int arow = tid >> 2;            // 0..63
  const int acol = (tid & 3) * 8;       // 0,8,16,24
  const int kp   = tid >> 4;            // 0..15 -> k pair 2*kp
  const int n0   = (tid & 15) * 8;      // 0..120

  const bf16_t* __restrict__ aPtr;      // this thread's A source chunk base
  {
    int g  = mbase + arow;
    int gc = (g < row1) ? g : (row1 - 1);
    aPtr = A + (size_t)gc * KDIM + acol;
  }
  const float* __restrict__ wPtr = Wr + (size_t)(2 * kp) * NDIM + nbase + n0;

  v8f acc[4] = {};

  uint4  aS;                   // staged A chunk (8 bf16)
  float4 w0a, w0b, w1a, w1b;   // staged W rows (2 k x 8 n fp32)

  // ---- prologue: load + stage K-tile 0 into buffer 0 ----
  aS  = *(const uint4*)(aPtr);
  w0a = *(const float4*)(wPtr);
  w0b = *(const float4*)(wPtr + 4);
  w1a = *(const float4*)(wPtr + NDIM);
  w1b = *(const float4*)(wPtr + NDIM + 4);
  {
    *(uint4*)(&sA[0][arow][acol]) = aS;
    union { bf16_t h[2]; unsigned int u; } pk;
    float lo[8] = {w0a.x, w0a.y, w0a.z, w0a.w, w0b.x, w0b.y, w0b.z, w0b.w};
    float hi[8] = {w1a.x, w1a.y, w1a.z, w1a.w, w1b.x, w1b.y, w1b.z, w1b.w};
#pragma unroll
    for (int u = 0; u < 8; ++u) {
      pk.h[0] = (bf16_t)lo[u];
      pk.h[1] = (bf16_t)hi[u];
      *(unsigned int*)(&sB[0][n0 + u][2 * kp]) = pk.u;
    }
  }
  __syncthreads();

  int cur = 0;
  for (int kk = 0; kk < KDIM; kk += BK) {
    const bool hasNext = (kk + BK) < KDIM;

    // ---- issue next K-tile's global loads first (overlap with WMMA) ----
    if (hasNext) {
      const int kn = kk + BK;
      aS  = *(const uint4*)(aPtr + kn);
      const float* __restrict__ w = wPtr + (size_t)kn * NDIM;
      w0a = *(const float4*)(w);
      w0b = *(const float4*)(w + 4);
      w1a = *(const float4*)(w + NDIM);
      w1b = *(const float4*)(w + NDIM + 4);
      if (kn + BK < KDIM) {
        __builtin_prefetch(w + (size_t)BK * NDIM, 0, 1);   // global_prefetch k+2
      }
    }

    // ---- batch all fragment loads, then issue WMMAs back-to-back ----
    union { v16bf v; uint4 q[2]; } afrag, bfrag[4];
    const bf16_t* ar = &sA[cur][wm * 16 + l15][0];
    afrag.q[0] = *(const uint4*)(ar + 8 * half);
    afrag.q[1] = *(const uint4*)(ar + 16 + 8 * half);
#pragma unroll
    for (int j = 0; j < 4; ++j) {
      const bf16_t* bc = &sB[cur][wn * 64 + j * 16 + l15][0];
      bfrag[j].q[0] = *(const uint4*)(bc + 8 * half);
      bfrag[j].q[1] = *(const uint4*)(bc + 16 + 8 * half);
    }
#pragma unroll
    for (int j = 0; j < 4; ++j) {
      acc[j] = __builtin_amdgcn_wmma_f32_16x16x32_bf16(
          false, afrag.v, false, bfrag[j].v, (short)0, acc[j], false, false);
    }

    // ---- convert + store next tile into the other buffer ----
    if (hasNext) {
      const int nxt = cur ^ 1;
      *(uint4*)(&sA[nxt][arow][acol]) = aS;
      union { bf16_t h[2]; unsigned int u; } pk;
      float lo[8] = {w0a.x, w0a.y, w0a.z, w0a.w, w0b.x, w0b.y, w0b.z, w0b.w};
      float hi[8] = {w1a.x, w1a.y, w1a.z, w1a.w, w1b.x, w1b.y, w1b.z, w1b.w};
#pragma unroll
      for (int u = 0; u < 8; ++u) {
        pk.h[0] = (bf16_t)lo[u];
        pk.h[1] = (bf16_t)hi[u];
        *(unsigned int*)(&sB[nxt][n0 + u][2 * kp]) = pk.u;
      }
    }
    __syncthreads();
    cur ^= 1;
  }

  // ---- epilogue ----
  const float* __restrict__ br = bias + (size_t)r * NDIM;
#pragma unroll
  for (int j = 0; j < 4; ++j) {
    const int col = nbase + wn * 64 + j * 16 + l15;
    const float bv = br[col];
#pragma unroll
    for (int i = 0; i < 8; ++i) {
      // D layout: VGPR i -> M = i (lanes 0-15) or 8+i (lanes 16-31); N = l15
      const int grow = mbase + wm * 16 + half * 8 + i;
      if (grow < row1) {
        float v = acc[j][i] + bv;
        if (RELU_BF16_OUT) {
          v = v > 0.0f ? v : 0.0f;
          outH[(size_t)grow * NDIM + col] = (bf16_t)v;
        } else {
          const int tok = list[grow];
          outL[(size_t)tok * NDIM + col] = v;
        }
      }
    }
  }
}

// ============ kernel 5: per-token argmax over V ============
__global__ __launch_bounds__(256) void cr_argmax(
    const float* __restrict__ logits, float* __restrict__ preds) {
  const int t = blockIdx.x;
  const float* __restrict__ p = logits + (size_t)t * VDIM;
  __shared__ float smax[256];
  __shared__ int   sidx[256];
  float best = -3.402823466e38f;
  int   bi   = 0;
  for (int v = threadIdx.x; v < VDIM; v += 256) {
    float x = p[v];
    if (x > best) { best = x; bi = v; }
  }
  smax[threadIdx.x] = best;
  sidx[threadIdx.x] = bi;
  __syncthreads();
  for (int s = 128; s > 0; s >>= 1) {
    if (threadIdx.x < s) {
      float o = smax[threadIdx.x + s];
      int  oi = sidx[threadIdx.x + s];
      float m = smax[threadIdx.x];
      if (o > m || (o == m && oi < sidx[threadIdx.x])) {
        smax[threadIdx.x] = o;
        sidx[threadIdx.x] = oi;
      }
    }
    __syncthreads();
  }
  if (threadIdx.x == 0) preds[t] = (float)sidx[0];
}

// ============ launcher ============
extern "C" void kernel_launch(void* const* d_in, const int* in_sizes, int n_in,
                              void* d_out, int out_size, void* d_ws, size_t ws_size,
                              hipStream_t stream) {
  const float* e   = (const float*)d_in[0];  // [B,S,D]
  const int*   rid = (const int*)  d_in[1];  // [B,S]
  const float* W1  = (const float*)d_in[2];  // [R,D,F]
  const float* b1  = (const float*)d_in[3];  // [R,F]
  const float* W2  = (const float*)d_in[4];  // [R,F,V]
  const float* b2  = (const float*)d_in[5];  // [R,V]

  float* logits = (float*)d_out;                 // [NTOK, V]
  float* preds  = logits + (size_t)NTOK * VDIM;  // [NTOK]

  char* ws = (char*)d_ws;
  int*    offs = (int*)(ws + WS_OFFS);
  int*    list = (int*)(ws + WS_LIST);
  bf16_t* Xg   = (bf16_t*)(ws + WS_XG);
  bf16_t* H    = (bf16_t*)(ws + WS_H);

  cr_build_lists<<<1, 256, 0, stream>>>(rid, offs, list);
  cr_gather_bf16<<<NTOK, 192, 0, stream>>>(e, list, Xg);

  // GEMM1: [n_r,768] x [768,3072] -> relu -> bf16 H
  cr_gemm_routed<DDIM, FDIM, true>
      <<<dim3(FDIM / 128, NTOK / 64, NROUTE), 256, 0, stream>>>(
          Xg, W1, b1, offs, nullptr, H, nullptr);

  // GEMM2: [n_r,3072] x [3072,1024] -> fp32 logits scattered to token rows
  cr_gemm_routed<FDIM, VDIM, false>
      <<<dim3(VDIM / 128, NTOK / 64, NROUTE), 256, 0, stream>>>(
          H, W2, b2, offs, list, nullptr, logits);

  cr_argmax<<<NTOK, 256, 0, stream>>>(logits, preds);
}
